// attn2d_31464930410820
// MI455X (gfx1250) — matrix operand order repaired and checked by reading, not compiled, not run on hardware
//
#include <hip/hip_runtime.h>
#include <hip/hip_bf16.h>

// ---------------------------------------------------------------------------
// MI455X (gfx1250) fused single-query 2D-rel attention block.
//
// Roofline: mandatory HBM traffic = memory tensor (411 MB) -> ~17.6us at
// 23.3 TB/s.  The naive k/v projections (210 GFLOP) are algebraically folded
// through the single query token down to ~16 GFLOP, all executed as
// v_wmma_f32_16x16x32_bf16.  Dense chain is batched 16-wide so the ~5 MB of
// weights stream from the 192 MB L2 once per 16 batches.
//
// memory[b] and U[b] are staged into LDS with GLOBAL_LOAD_ASYNC_TO_LDS_B128
// (ASYNCcnt + s_wait_asynccnt) instead of a VGPR round-trip; k3 prefetches
// the next weight panel with global_prefetch.
//
// Workspace use: U/ctx (B*4096 bf16 = 32 MB) + transposed bf16 weights
// (~2.4 MB)  => requires ws_size >= ~35 MB.
// ---------------------------------------------------------------------------

typedef __bf16 bf16_t;
typedef __attribute__((ext_vector_type(16))) __bf16 v16bf;
typedef __attribute__((ext_vector_type(8)))  float  v8f;

#define BATCH   4096
#define HWDIM   49
#define CDIM    512
#define NHEAD   8
#define QSCALE  0.125f               // DKH^-0.5 = 64^-0.5
#define PRED_BASE ((size_t)BATCH * 512)   // out2 comes first in d_out

#if __has_builtin(__builtin_amdgcn_global_load_async_to_lds_b128)
#define HAVE_ASYNC_LDS 1
// builtin signature (from hipcc diagnostic): (int4 AS1* src, int4 AS3* dst, Ii, Ii)
typedef int v4i_t __attribute__((vector_size(16)));
typedef __attribute__((address_space(1))) v4i_t gv4i_t;
typedef __attribute__((address_space(3))) v4i_t lv4i_t;
#endif

__device__ __forceinline__ void wait_async0() {
#if defined(HAVE_ASYNC_LDS)
#if __has_builtin(__builtin_amdgcn_s_wait_asynccnt)
  __builtin_amdgcn_s_wait_asynccnt(0);
#else
  asm volatile("s_wait_asynccnt 0" ::: "memory");
#endif
#endif
}

__device__ __forceinline__ v8f wmma_bf16(v16bf a, v16bf b, v8f c) {
  // D = A(16x32) * B(32x16) + C, f32 accumulate
  return __builtin_amdgcn_wmma_f32_16x16x32_bf16(false, a, false, b,
                                                 (short)0, c, false, false);
}

// --- A-tile loader: 16(M) x 32(K) bf16 from row-major [row][ld], row == M.
// Layout: lanes 0-15 hold K 0..15, lanes 16-31 hold K 16..31 (matches B).
__device__ __forceinline__ v16bf load_a_bf16(const bf16_t* base, int ld, int k0,
                                             int nrows) {
  const int lane = threadIdx.x & 31;
  const int r = lane & 15;
  const int kb = k0 + ((lane & 16) ? 16 : 0);
  v16bf o;
  if (r < nrows) {
    const bf16_t* p = base + (size_t)r * ld + kb;
#pragma unroll
    for (int i = 0; i < 16; ++i) o[i] = p[i];
  } else {
#pragma unroll
    for (int i = 0; i < 16; ++i) o[i] = (bf16_t)0.0f;
  }
  return o;
}

__device__ __forceinline__ v16bf load_a_f32(const float* base, int ld, int k0,
                                            int nrows) {
  const int lane = threadIdx.x & 31;
  const int r = lane & 15;
  const int kb = k0 + ((lane & 16) ? 16 : 0);
  v16bf o;
  if (r < nrows) {
    const float* p = base + (size_t)r * ld + kb;
#pragma unroll
    for (int i = 0; i < 16; ++i) o[i] = (bf16_t)p[i];
  } else {
#pragma unroll
    for (int i = 0; i < 16; ++i) o[i] = (bf16_t)0.0f;
  }
  return o;
}

// --- B-tile loader: 32(K) x 16(N) bf16 from row-major [n][ld] (i.e. W^T).
// Contiguous-k layout used here (same as A): lanes 0-15 hold K 0..15, lanes 16-31 hold K 16..31.
__device__ __forceinline__ v16bf load_b_bf16(const bf16_t* base, int ld, int k0,
                                             int nrows) {
  const int lane = threadIdx.x & 31;
  const int n = lane & 15;
  const int kb = k0 + ((lane & 16) ? 16 : 0);
  v16bf o;
  if (n < nrows) {
    const bf16_t* p = base + (size_t)n * ld + kb;
#pragma unroll
    for (int i = 0; i < 16; ++i) o[i] = p[i];
  } else {
#pragma unroll
    for (int i = 0; i < 16; ++i) o[i] = (bf16_t)0.0f;
  }
  return o;
}

__device__ __forceinline__ v16bf load_b_f32(const float* base, int ld, int k0,
                                            int nrows) {
  const int lane = threadIdx.x & 31;
  const int n = lane & 15;
  const int kb = k0 + ((lane & 16) ? 16 : 0);
  v16bf o;
  if (n < nrows) {
    const float* p = base + (size_t)n * ld + kb;
#pragma unroll
    for (int i = 0; i < 16; ++i) o[i] = (bf16_t)p[i];
  } else {
#pragma unroll
    for (int i = 0; i < 16; ++i) o[i] = (bf16_t)0.0f;
  }
  return o;
}

// --- B-tile loader from a K-major f32 matrix [k][ld] (n along the row).
// For each element index i, the 32 lanes read 16 consecutive dwords: no LDS
// bank conflicts.
__device__ __forceinline__ v16bf load_b_f32_km(const float* base, int ld, int k0) {
  const int lane = threadIdx.x & 31;
  const int n = lane & 15;
  const int kb = k0 + ((lane & 16) ? 16 : 0);
  v16bf o;
#pragma unroll
  for (int i = 0; i < 16; ++i) o[i] = (bf16_t)base[(size_t)(kb + i) * ld + n];
  return o;
}

// ---------------------------------------------------------------------------
// k0: convert W (KxN f32, row-major) -> W^T (Npad x K bf16, row-major),
// zero-padding rows n in [N, Npad).
// ---------------------------------------------------------------------------
__global__ void k0_transpose(const float* __restrict__ src, bf16_t* __restrict__ dst,
                             int K, int N, int Npad) {
  int idx = blockIdx.x * blockDim.x + threadIdx.x;
  if (idx >= Npad * K) return;
  int n = idx / K;
  int k = idx - n * K;
  dst[idx] = (n < N) ? (bf16_t)src[(size_t)k * N + n] : (bf16_t)0.0f;
}

// ---------------------------------------------------------------------------
// k1: fold scaled q through W_k per head:
//   U[b, h*512+c] = sum_d qs[b,h*64+d] * W_k[c, h*64+d]
// 16 batches per WG, WMMA M=16 x N=4096 x K=64(per head block).
// ---------------------------------------------------------------------------
__global__ __launch_bounds__(256) void k1_foldq(const float* __restrict__ q,
                                                const float* __restrict__ Wk,
                                                bf16_t* __restrict__ U) {
  __shared__ bf16_t qs[16][512];
  const int b0 = blockIdx.x * 16;
  for (int idx = threadIdx.x; idx < 16 * 512; idx += 256) {
    int m = idx >> 9, d = idx & 511;
    qs[m][d] = (bf16_t)(q[(size_t)(b0 + m) * 512 + d] * QSCALE);
  }
  __syncthreads();
  const int wave = threadIdx.x >> 5;
  const int lane = threadIdx.x & 31;
  for (int t = wave; t < 256; t += 8) {        // 256 N-tiles over 4096 cols
    const int n0 = t * 16;
    const int h = n0 >> 9;                     // head of this U column block
    const int c0 = n0 & 511;
    v8f acc = {0.f, 0.f, 0.f, 0.f, 0.f, 0.f, 0.f, 0.f};
#pragma unroll
    for (int k0 = 0; k0 < 64; k0 += 32) {
      v16bf a = load_a_bf16(&qs[0][h * 64], 512, k0, 16);
      // B[k=d][n=c] = W_k[c0+n, h*64+d]; W_k rows are already [n][k] order.
      v16bf b = load_b_f32(Wk + (size_t)c0 * 512 + h * 64, 512, k0, 16);
      acc = wmma_bf16(a, b, acc);
    }
    const int n = lane & 15;
    const int mb = (lane & 16) ? 8 : 0;
#pragma unroll
    for (int vr = 0; vr < 8; ++vr) {
      int m = mb + vr;
      U[(size_t)(b0 + m) * 4096 + n0 + n] = (bf16_t)acc[vr];
    }
  }
}

// ---------------------------------------------------------------------------
// k2: per-batch attention.  One WG per batch.
//   memory[b] (49x512 f32, 100 KB) async-DMA'd straight into LDS (ASYNCcnt),
//   U[b] (8 KB bf16) likewise.
//   logits[j,h] = mem[j,:] . U[b,h,:]           (WMMA, M=64 N=16(8 used) K=512)
//   logits += rel_logits + prev_attn; softmax over j
//   ctx[h,c]  = sum_j w[h,j] mem[j,c]           (WMMA, M=16(8) N=512 K=64)
// ctx overwrites U[b] in place (bf16).
// ---------------------------------------------------------------------------
__global__ __launch_bounds__(256) void k2_attn(const float* __restrict__ memory,
                                               const float* __restrict__ q,
                                               const float* __restrict__ relh,
                                               const float* __restrict__ relw,
                                               const float* __restrict__ prev,
                                               bf16_t* __restrict__ Uctx) {
  __shared__ float  memF[64][512];   // mem[j][c] f32, rows >= 49 zeroed
  __shared__ bf16_t u_s[8][512];     // U[b][h][c] bf16 (logits B side)
  __shared__ float  lg[64][16];      // raw logits [j][h]
  __shared__ float  rp[8][64];       // rel + prev [h][j]
  __shared__ bf16_t wts[16][64];     // softmax weights [h][j], padded
  __shared__ float  qs[512];         // scaled q

  const int b = blockIdx.x;
  const int tid = threadIdx.x;

#if defined(HAVE_ASYNC_LDS)
  {
    // memory[b]: 49*512*4 B = 6272 16-byte lane transfers, global -> LDS.
    const char* gsrc = (const char*)(memory + (size_t)b * HWDIM * 512);
    char* ldst = (char*)&memF[0][0];
    for (int idx = tid; idx < (HWDIM * 512 * 4) / 16; idx += 256)
      __builtin_amdgcn_global_load_async_to_lds_b128(
          (gv4i_t*)(gsrc + (size_t)idx * 16), (lv4i_t*)(ldst + (size_t)idx * 16),
          0, 0);
    // U[b]: 8*512*2 B = 512 lane transfers.
    const char* gsu = (const char*)(Uctx + (size_t)b * 4096);
    char* ldu = (char*)&u_s[0][0];
    for (int idx = tid; idx < (NHEAD * 512 * 2) / 16; idx += 256)
      __builtin_amdgcn_global_load_async_to_lds_b128(
          (gv4i_t*)(gsu + (size_t)idx * 16), (lv4i_t*)(ldu + (size_t)idx * 16),
          0, 0);
  }
#else
  for (int idx = tid; idx < HWDIM * 512; idx += 256)
    memF[idx >> 9][idx & 511] = memory[(size_t)b * HWDIM * 512 + idx];
  for (int idx = tid; idx < NHEAD * 512; idx += 256)
    u_s[idx >> 9][idx & 511] = Uctx[(size_t)b * 4096 + idx];
#endif
  // zero pad rows j in [49, 64) (disjoint from the async-written region)
  for (int idx = tid; idx < (64 - HWDIM) * 512; idx += 256)
    memF[HWDIM + (idx >> 9)][idx & 511] = 0.0f;
  for (int idx = tid; idx < 512; idx += 256)
    qs[idx] = q[(size_t)b * 512 + idx] * QSCALE;
  wait_async0();
  __syncthreads();

  const int wave = tid >> 5;
  const int lane = tid & 31;

  if (wave < 4) {
    // logits GEMM, m-tile = wave (j rows 16*wave .. +15)
    v8f acc = {0.f, 0.f, 0.f, 0.f, 0.f, 0.f, 0.f, 0.f};
    for (int k0 = 0; k0 < 512; k0 += 32) {
      v16bf a = load_a_f32(&memF[wave * 16][0], 512, k0, 16);
      v16bf bm = load_b_bf16(&u_s[0][0], 512, k0, NHEAD);
      acc = wmma_bf16(a, bm, acc);
    }
    const int n = lane & 15;
    const int mb = (lane & 16) ? 8 : 0;
#pragma unroll
    for (int vr = 0; vr < 8; ++vr) lg[wave * 16 + mb + vr][n] = acc[vr];
  } else {
    // rel_logits + prev_attn for two heads per wave
    for (int hh = 0; hh < 2; ++hh) {
      int h = (wave - 4) * 2 + hh;
      for (int j = lane; j < 64; j += 32) {
        float s = 0.0f;
        if (j < HWDIM) {
          s = prev[(size_t)b * HWDIM + j];
          const float* rh = relh + (size_t)j * 256 + h * 32;
          const float* rw = relw + (size_t)j * 256 + h * 32;
          const float* qh = qs + h * 64;
#pragma unroll
          for (int d = 0; d < 32; ++d) s += rh[d] * qh[d] + rw[d] * qh[32 + d];
        }
        rp[h][j] = s;
      }
    }
  }
  __syncthreads();

  // softmax over j (49 valid), wave h handles head h
  {
    const int h = wave;
    const int j1 = lane + 32;
    float x0 = lg[lane][h] + rp[h][lane];             // lane < 32 < 49: valid
    float x1 = (j1 < HWDIM) ? (lg[j1][h] + rp[h][j1]) : -__builtin_inff();
    float mx = fmaxf(x0, x1);
#pragma unroll
    for (int m = 16; m >= 1; m >>= 1) mx = fmaxf(mx, __shfl_xor(mx, m, 32));
    float e0 = __expf(x0 - mx);
    float e1 = (j1 < HWDIM) ? __expf(x1 - mx) : 0.0f;
    float sm = e0 + e1;
#pragma unroll
    for (int m = 16; m >= 1; m >>= 1) sm += __shfl_xor(sm, m, 32);
    float inv = 1.0f / sm;
    wts[h][lane] = (bf16_t)(e0 * inv);
    wts[h][j1] = (bf16_t)(e1 * inv);
    wts[h + 8][lane] = (bf16_t)0.0f;     // pad heads 8..15 with zeros
    wts[h + 8][j1] = (bf16_t)0.0f;
  }
  __syncthreads();

  // ctx GEMM: ctx[h, c] = sum_j w[h,j] * mem[j,c]; write bf16 over U[b]
  for (int t = wave; t < 32; t += 8) {
    const int n0 = t * 16;
    v8f acc = {0.f, 0.f, 0.f, 0.f, 0.f, 0.f, 0.f, 0.f};
#pragma unroll
    for (int k0 = 0; k0 < 64; k0 += 32) {
      v16bf a = load_a_bf16(&wts[0][0], 64, k0, 16);
      v16bf bm = load_b_f32_km(&memF[0][n0], 512, k0);  // B[k=j][n=c]
      acc = wmma_bf16(a, bm, acc);
    }
    const int n = lane & 15;
    const int mb = (lane & 16) ? 8 : 0;
#pragma unroll
    for (int vr = 0; vr < 8; ++vr) {
      int m = mb + vr;
      if (m < NHEAD)
        Uctx[(size_t)b * 4096 + (size_t)m * 512 + n0 + n] = (bf16_t)acc[vr];
    }
  }
}

// ---------------------------------------------------------------------------
// k3: batched dense chain, 16 batches per WG.  All matmuls via WMMA against
// pre-transposed bf16 weights (B-side directly from L2-resident workspace,
// with global_prefetch hints one panel ahead).
// ---------------------------------------------------------------------------
__global__ __launch_bounds__(256) void k3_dense(
    const bf16_t* __restrict__ ctx, const float* __restrict__ q,
    const bf16_t* __restrict__ WvT, const bf16_t* __restrict__ WzT,
    const bf16_t* __restrict__ F1T, const bf16_t* __restrict__ F2T,
    const bf16_t* __restrict__ P1T, const bf16_t* __restrict__ P2T,
    const float* __restrict__ ln1g, const float* __restrict__ ln1b,
    const float* __restrict__ fb1, const float* __restrict__ fb2,
    const float* __restrict__ ln2g, const float* __restrict__ ln2b,
    float* __restrict__ out) {
  __shared__ float  z_f[16][512];     // z (then reused as h2)
  __shared__ float  o1_f[16][512];    // out1 f32
  __shared__ bf16_t o1_bf[16][512];   // out1 bf16 (A side of fc1)
  __shared__ char   bufA[16 * 512 * 2];  // attn_bf -> p1_bf -> h1_bf
  __shared__ char   bufB[16 * 512 * 2];  // z_bf -> p2_f
  bf16_t (*attn_bf)[512] = (bf16_t(*)[512])bufA;
  bf16_t (*p1_bf)[256]   = (bf16_t(*)[256])bufA;
  bf16_t (*h1_bf)[512]   = (bf16_t(*)[512])bufA;
  bf16_t (*z_bf)[512]    = (bf16_t(*)[512])bufB;
  float  (*p2_f)[128]    = (float(*)[128])bufB;

  const int b0 = blockIdx.x * 16;
  const int wave = threadIdx.x >> 5;
  const int lane = threadIdx.x & 31;
  const int n = lane & 15;
  const int mb = (lane & 16) ? 8 : 0;

  // ---- attn = per-head ctx @ W_v  (A straight from global ctx bf16) ----
  for (int t = wave; t < 32; t += 8) {
    const int n0 = t * 16;
    const int h = n0 >> 6;              // head owning these 16 attn columns
    if (t + 8 < 32) __builtin_prefetch(WvT + (size_t)(n0 + 128) * 512, 0, 1);
    v8f acc = {0.f, 0.f, 0.f, 0.f, 0.f, 0.f, 0.f, 0.f};
    for (int k0 = 0; k0 < 512; k0 += 32) {
      v16bf a = load_a_bf16(ctx + (size_t)b0 * 4096 + h * 512, 4096, k0, 16);
      v16bf bm = load_b_bf16(WvT + (size_t)n0 * 512, 512, k0, 16);
      acc = wmma_bf16(a, bm, acc);
    }
#pragma unroll
    for (int vr = 0; vr < 8; ++vr) attn_bf[mb + vr][n0 + n] = (bf16_t)acc[vr];
  }
  __syncthreads();

  // ---- z = attn @ W_z ----
  for (int t = wave; t < 32; t += 8) {
    const int n0 = t * 16;
    if (t + 8 < 32) __builtin_prefetch(WzT + (size_t)(n0 + 128) * 512, 0, 1);
    v8f acc = {0.f, 0.f, 0.f, 0.f, 0.f, 0.f, 0.f, 0.f};
    for (int k0 = 0; k0 < 512; k0 += 32) {
      v16bf a = load_a_bf16(&attn_bf[0][0], 512, k0, 16);
      v16bf bm = load_b_bf16(WzT + (size_t)n0 * 512, 512, k0, 16);
      acc = wmma_bf16(a, bm, acc);
    }
#pragma unroll
    for (int vr = 0; vr < 8; ++vr) {
      float v = acc[vr];
      z_f[mb + vr][n0 + n] = v;
      z_bf[mb + vr][n0 + n] = (bf16_t)v;
    }
  }
  __syncthreads();

  // ---- pred1 = relu(z @ pred_w1)  (N=256) ----
  for (int t = wave; t < 16; t += 8) {
    const int n0 = t * 16;
    v8f acc = {0.f, 0.f, 0.f, 0.f, 0.f, 0.f, 0.f, 0.f};
    for (int k0 = 0; k0 < 512; k0 += 32) {
      v16bf a = load_a_bf16(&z_bf[0][0], 512, k0, 16);
      v16bf bm = load_b_bf16(P1T + (size_t)n0 * 512, 512, k0, 16);
      acc = wmma_bf16(a, bm, acc);
    }
#pragma unroll
    for (int vr = 0; vr < 8; ++vr)
      p1_bf[mb + vr][n0 + n] = (bf16_t)fmaxf(acc[vr], 0.0f);
  }
  __syncthreads();

  // ---- pred2 = pred1 @ pred_w2  (N=128 padded, 121 valid) ----
  for (int t = wave; t < 8; t += 8) {
    const int n0 = t * 16;
    v8f acc = {0.f, 0.f, 0.f, 0.f, 0.f, 0.f, 0.f, 0.f};
#pragma unroll
    for (int k0 = 0; k0 < 256; k0 += 32) {
      v16bf a = load_a_bf16(&p1_bf[0][0], 256, k0, 16);
      v16bf bm = load_b_bf16(P2T + (size_t)n0 * 256, 256, k0, 16);
      acc = wmma_bf16(a, bm, acc);
    }
#pragma unroll
    for (int vr = 0; vr < 8; ++vr) p2_f[mb + vr][n0 + n] = acc[vr];
  }
  __syncthreads();

  // ---- write attn_pred: (11x11) padded by 4 into (15x15) ----
  for (int idx = threadIdx.x; idx < 16 * 225; idx += 256) {
    int m = idx / 225, t = idx - m * 225;
    int r = t / 15, cc = t - r * 15;
    float v = 0.0f;
    if (r >= 4 && cc >= 4) v = p2_f[m][(r - 4) * 11 + (cc - 4)];
    out[PRED_BASE + (size_t)(b0 + m) * 225 + t] = v;
  }

  // ---- out1 = LN(z + residual); residual = q * scale (aliased in-place) ----
  for (int rr = wave * 2; rr < wave * 2 + 2; ++rr) {
    float xs[16], s = 0.0f, s2 = 0.0f;
#pragma unroll
    for (int i = 0; i < 16; ++i) {
      int c = lane + i * 32;
      float x = z_f[rr][c] + q[(size_t)(b0 + rr) * 512 + c] * QSCALE;
      xs[i] = x; s += x; s2 += x * x;
    }
#pragma unroll
    for (int m = 16; m >= 1; m >>= 1) {
      s += __shfl_xor(s, m, 32);
      s2 += __shfl_xor(s2, m, 32);
    }
    float mean = s * (1.0f / 512.0f);
    float var = s2 * (1.0f / 512.0f) - mean * mean;
    float inv = rsqrtf(var + 1e-5f);
#pragma unroll
    for (int i = 0; i < 16; ++i) {
      int c = lane + i * 32;
      float y = (xs[i] - mean) * inv * ln1g[c] + ln1b[c];
      o1_f[rr][c] = y;
      o1_bf[rr][c] = (bf16_t)y;
    }
  }
  __syncthreads();

  // ---- h1 = relu(out1 @ fc_w1 + b1) ----
  for (int t = wave; t < 32; t += 8) {
    const int n0 = t * 16;
    if (t + 8 < 32) __builtin_prefetch(F1T + (size_t)(n0 + 128) * 512, 0, 1);
    v8f acc = {0.f, 0.f, 0.f, 0.f, 0.f, 0.f, 0.f, 0.f};
    for (int k0 = 0; k0 < 512; k0 += 32) {
      v16bf a = load_a_bf16(&o1_bf[0][0], 512, k0, 16);
      v16bf bm = load_b_bf16(F1T + (size_t)n0 * 512, 512, k0, 16);
      acc = wmma_bf16(a, bm, acc);
    }
#pragma unroll
    for (int vr = 0; vr < 8; ++vr)
      h1_bf[mb + vr][n0 + n] = (bf16_t)fmaxf(acc[vr] + fb1[n0 + n], 0.0f);
  }
  __syncthreads();

  // ---- h2 = h1 @ fc_w2 + b2  (into z_f, which is dead) ----
  for (int t = wave; t < 32; t += 8) {
    const int n0 = t * 16;
    if (t + 8 < 32) __builtin_prefetch(F2T + (size_t)(n0 + 128) * 512, 0, 1);
    v8f acc = {0.f, 0.f, 0.f, 0.f, 0.f, 0.f, 0.f, 0.f};
    for (int k0 = 0; k0 < 512; k0 += 32) {
      v16bf a = load_a_bf16(&h1_bf[0][0], 512, k0, 16);
      v16bf bm = load_b_bf16(F2T + (size_t)n0 * 512, 512, k0, 16);
      acc = wmma_bf16(a, bm, acc);
    }
#pragma unroll
    for (int vr = 0; vr < 8; ++vr) z_f[mb + vr][n0 + n] = acc[vr] + fb2[n0 + n];
  }
  __syncthreads();

  // ---- out2 = LN(h2 + out1) -> d_out ----
  for (int rr = wave * 2; rr < wave * 2 + 2; ++rr) {
    float xs[16], s = 0.0f, s2 = 0.0f;
#pragma unroll
    for (int i = 0; i < 16; ++i) {
      int c = lane + i * 32;
      float x = z_f[rr][c] + o1_f[rr][c];
      xs[i] = x; s += x; s2 += x * x;
    }
#pragma unroll
    for (int m = 16; m >= 1; m >>= 1) {
      s += __shfl_xor(s, m, 32);
      s2 += __shfl_xor(s2, m, 32);
    }
    float mean = s * (1.0f / 512.0f);
    float var = s2 * (1.0f / 512.0f) - mean * mean;
    float inv = rsqrtf(var + 1e-5f);
#pragma unroll
    for (int i = 0; i < 16; ++i) {
      int c = lane + i * 32;
      out[(size_t)(b0 + rr) * 512 + c] =
          (xs[i] - mean) * inv * ln2g[c] + ln2b[c];
    }
  }
}

// ---------------------------------------------------------------------------
extern "C" void kernel_launch(void* const* d_in, const int* in_sizes, int n_in,
                              void* d_out, int out_size, void* d_ws, size_t ws_size,
                              hipStream_t stream) {
  (void)in_sizes; (void)n_in; (void)out_size; (void)ws_size;
  const float* q      = (const float*)d_in[0];
  const float* memory = (const float*)d_in[1];
  const float* rel_h  = (const float*)d_in[2];
  const float* rel_w  = (const float*)d_in[3];
  const float* prev   = (const float*)d_in[4];
  const float* W_k    = (const float*)d_in[5];
  const float* W_v    = (const float*)d_in[6];
  const float* W_z    = (const float*)d_in[7];
  const float* ln1g   = (const float*)d_in[8];
  const float* ln1b   = (const float*)d_in[9];
  const float* fc_w1  = (const float*)d_in[10];
  const float* fc_b1  = (const float*)d_in[11];
  const float* fc_w2  = (const float*)d_in[12];
  const float* fc_b2  = (const float*)d_in[13];
  const float* ln2g   = (const float*)d_in[14];
  const float* ln2b   = (const float*)d_in[15];
  const float* p_w1   = (const float*)d_in[16];
  const float* p_w2   = (const float*)d_in[17];
  float* out = (float*)d_out;

  bf16_t* U   = (bf16_t*)d_ws;                       // B x 4096 (U, then ctx)
  bf16_t* WT  = U + (size_t)BATCH * 4096;
  bf16_t* WvT = WT;                                  // 512x512
  bf16_t* WzT = WT + 262144;                         // 512x512
  bf16_t* F1T = WT + 524288;                         // 512x512
  bf16_t* F2T = WT + 786432;                         // 512x512
  bf16_t* P1T = WT + 1048576;                        // 256x512
  bf16_t* P2T = WT + 1179648;                        // 128x256 (121 valid)

  k0_transpose<<<(512 * 512 + 255) / 256, 256, 0, stream>>>(W_v, WvT, 512, 512, 512);
  k0_transpose<<<(512 * 512 + 255) / 256, 256, 0, stream>>>(W_z, WzT, 512, 512, 512);
  k0_transpose<<<(512 * 512 + 255) / 256, 256, 0, stream>>>(fc_w1, F1T, 512, 512, 512);
  k0_transpose<<<(512 * 512 + 255) / 256, 256, 0, stream>>>(fc_w2, F2T, 512, 512, 512);
  k0_transpose<<<(256 * 512 + 255) / 256, 256, 0, stream>>>(p_w1, P1T, 512, 256, 256);
  k0_transpose<<<(128 * 256 + 255) / 256, 256, 0, stream>>>(p_w2, P2T, 256, 121, 128);

  k1_foldq<<<BATCH / 16, 256, 0, stream>>>(q, W_k, U);
  k2_attn<<<BATCH, 256, 0, stream>>>(memory, q, rel_h, rel_w, prev, U);
  k3_dense<<<BATCH / 16, 256, 0, stream>>>(U, q, WvT, WzT, F1T, F2T, P1T, P2T,
                                           ln1g, ln1b, fc_b1, fc_b2, ln2g, ln2b,
                                           out);
}
